// MGAT_56092272885968
// MI455X (gfx1250) — matrix operand order repaired and checked by reading, not compiled, not run on hardware
//
#include <hip/hip_runtime.h>

#define Nn   256
#define Ss   64
#define Ww   256
#define Hh   32
#define Gg   96          // 3*H
#define DIN  768         // 3*W
#define ALPHAC 0.2f
#define NEGV  -9.0e15f

typedef unsigned short u16;
typedef __bf16 bf16_t;
typedef __attribute__((ext_vector_type(16))) bf16_t         v16bf;
typedef __attribute__((ext_vector_type(8)))  float          v8f;
typedef __attribute__((ext_vector_type(8)))  unsigned int   v8u;
typedef __attribute__((ext_vector_type(4)))  unsigned int   v4u;

#define WMMA_BF16(A,B,C) __builtin_amdgcn_wmma_f32_16x16x32_bf16(false,(A),false,(B),(short)0,(C),false,false)

__device__ __forceinline__ u16 f2bf(float f) {
  unsigned int u = __builtin_bit_cast(unsigned int, f);
  unsigned int r = u + 0x7FFFu + ((u >> 16) & 1u);   // RNE
  return (u16)(r >> 16);
}
__device__ __forceinline__ float bf2f(u16 b) {
  unsigned int u = ((unsigned int)b) << 16;
  return __builtin_bit_cast(float, u);
}
__device__ __forceinline__ unsigned int pack2bf(float lo, float hi) {
  return (unsigned int)f2bf(lo) | ((unsigned int)f2bf(hi) << 16);
}
__device__ __forceinline__ v8f vzero8() {
  v8f z;
#pragma unroll
  for (int i = 0; i < 8; ++i) z[i] = 0.f;
  return z;
}

// ---- WMMA-fragment-swizzled LDS tiles: 16x32 (A) / 32x16 (B), 512 bf16 = 1KB each.
// Within a tile, lane L's 16 elements are contiguous at offset L*16 -> frag load = 2x ds_load_b128.
__device__ __forceinline__ int a_idx(int numKT, int m, int k) {        // element (m,k), A-swizzled
  int t    = ((m >> 4) * numKT + (k >> 5)) << 9;
  int lane = (m & 15) + (((k >> 4) & 1) << 4);
  int e    = k & 15;
  return t + (lane << 4) + e;
}
__device__ __forceinline__ int b_idx(int numNT, int k, int n) {        // element (k,n), B-swizzled
  int t    = ((k >> 5) * numNT + (n >> 4)) << 9;
  int lane = (n & 15) + (((k >> 4) & 1) << 4);
  int e    = k & 15;
  return t + (lane << 4) + e;
}
__device__ __forceinline__ v16bf load_frag(const u16* tile, int lane) {
  const v4u* p = (const v4u*)(tile + (lane << 4));
  v4u lo = p[0], hi = p[1];
  v8u r = __builtin_shufflevector(lo, hi, 0, 1, 2, 3, 4, 5, 6, 7);
  return __builtin_bit_cast(v16bf, r);
}

// ---- staging: f32 global -> bf16 swizzled LDS, packed 32-bit stores (pairs along k).
__device__ __forceinline__ void stage_a_rm(u16* sA, const float* g, int M, int K, int ldg,
                                           int numKT, int tid, int nthr) {
  unsigned int* s32 = (unsigned int*)sA;
  int K2 = K >> 1;
  for (int i = tid; i < M * K2; i += nthr) {
    int m = i / K2, k = (i - m * K2) << 1;
    s32[a_idx(numKT, m, k) >> 1] = pack2bf(g[m * ldg + k], g[m * ldg + k + 1]);
  }
}
__device__ __forceinline__ void stage_a_tr(u16* sA, const float* g, int M, int K, int ldg,
                                           int numKT, int tid, int nthr) {   // A[m][k]=g[k*ldg+m]
  unsigned int* s32 = (unsigned int*)sA;
  int K2 = K >> 1;
  for (int i = tid; i < M * K2; i += nthr) {
    int m = i / K2, k = (i - m * K2) << 1;
    s32[a_idx(numKT, m, k) >> 1] = pack2bf(g[k * ldg + m], g[(k + 1) * ldg + m]);
  }
}
__device__ __forceinline__ void stage_b_rm(u16* sB, const float* g, int K, int N, int ldg,
                                           int numNT, int tid, int nthr) {
  unsigned int* s32 = (unsigned int*)sB;
  int K2 = K >> 1;
  for (int i = tid; i < K2 * N; i += nthr) {
    int k2 = i / N, n = i - k2 * N, k = k2 << 1;
    s32[b_idx(numNT, k, n) >> 1] = pack2bf(g[k * ldg + n], g[(k + 1) * ldg + n]);
  }
}
__device__ __forceinline__ void stage_b_tr(u16* sB, const float* g, int K, int N, int ldg,
                                           int numNT, int tid, int nthr) {   // B[k][n]=g[n*ldg+k]
  unsigned int* s32 = (unsigned int*)sB;
  int K2 = K >> 1;
  for (int i = tid; i < K2 * N; i += nthr) {
    int k2 = i / N, n = i - k2 * N, k = k2 << 1;
    s32[b_idx(numNT, k, n) >> 1] = pack2bf(g[n * ldg + k], g[n * ldg + k + 1]);
  }
}

// B fragment assembled from GLOBAL f32 transposed source: B[k][col] = M[col][k] (16 consec floats/lane).
__device__ __forceinline__ v16bf load_bT_g(const float* M_, int ldk, int lane) {
  int col = lane & 15;
  int kb  = (lane >> 4) << 4;
  const float* p = M_ + col * ldk + kb;
  v8u b;
#pragma unroll
  for (int v = 0; v < 8; ++v) b[v] = pack2bf(p[2 * v], p[2 * v + 1]);
  return __builtin_bit_cast(v16bf, b);
}

// ---- accumulator stores
__device__ __forceinline__ void store_c_g(float* base, int ld, int lane, v8f c) {
  int col = lane & 15, r0 = (lane >> 4) << 3;
#pragma unroll
  for (int v = 0; v < 8; ++v) base[(r0 + v) * ld + col] = c[v];
}
__device__ __forceinline__ void store_c_gT(float* base, int ldT, int lane, v8f c) {
  int col = lane & 15, r0 = (lane >> 4) << 3;
#pragma unroll
  for (int v = 0; v < 8; ++v) base[col * ldT + r0 + v] = c[v];
}
__device__ __forceinline__ void store_c_bias_g(float* base, int ld, const float* bias, int lane, v8f c) {
  int col = lane & 15, r0 = (lane >> 4) << 3;
  float b = bias[col];
#pragma unroll
  for (int v = 0; v < 8; ++v) base[(r0 + v) * ld + col] = c[v] + b;
}
// C tile -> B-swizzled LDS (output rows become next GEMM's K): one 16B ds store per lane.
__device__ __forceinline__ void store_c_bswz(u16* sB, int numNT, int m0, int n0, int lane, v8f c) {
  int col = lane & 15, r0 = (lane >> 4) << 3;
  int idx = b_idx(numNT, m0 + r0, n0 + col);     // 8 contiguous u16, 16B aligned
  v4u w;
  w[0] = pack2bf(c[0], c[1]); w[1] = pack2bf(c[2], c[3]);
  w[2] = pack2bf(c[4], c[5]); w[3] = pack2bf(c[6], c[7]);
  *(v4u*)(sB + idx) = w;
}
// C tile -> A-swizzled LDS (output is next GEMM's A): scattered u16 stores.
__device__ __forceinline__ void store_c_aswz(u16* sA, int numKT, int m0, int k0, int lane, v8f c) {
  int col = lane & 15, r0 = (lane >> 4) << 3;
  int k = k0 + col;
#pragma unroll
  for (int v = 0; v < 8; ++v) sA[a_idx(numKT, m0 + r0 + v, k)] = f2bf(c[v]);
}
__device__ __forceinline__ void store_c_aswz_relu_bias(u16* sA, int numKT, int m0, int k0,
                                                       const float* bias, int lane, v8f c) {
  int col = lane & 15, r0 = (lane >> 4) << 3;
  int k = k0 + col;
  float b = bias[col];
#pragma unroll
  for (int v = 0; v < 8; ++v) {
    float t = c[v] + b;
    sA[a_idx(numKT, m0 + r0 + v, k)] = f2bf(t > 0.f ? t : 0.f);
  }
}
__device__ __forceinline__ float read_bswz(const u16* sB, int numNT, int k, int n) {
  return bf2f(sB[b_idx(numNT, k, n)]);
}
__device__ __forceinline__ float read_aswz(const u16* sA, int numKT, int m, int k) {
  return bf2f(sA[a_idx(numKT, m, k)]);
}

// Masked-softmax row by one wave32; result written directly in A-swizzled layout.
__device__ __forceinline__ void att_row_aswz(u16* sA, int numKT, int r, const float* adjRow,
                                             float f1i, const float* f2s, int per, int lane) {
  float vals[8];
  float mx = -3.0e38f;
  for (int i = 0; i < per; ++i) {
    int j = lane + (i << 5);
    float e = f1i + f2s[j];
    e = (e >= 0.f) ? e : ALPHAC * e;
    float v = (adjRow[j] > 0.f) ? e : NEGV;
    vals[i] = v;
    mx = fmaxf(mx, v);
  }
#pragma unroll
  for (int o = 16; o >= 1; o >>= 1) mx = fmaxf(mx, __shfl_xor(mx, o, 32));
  float sum = 0.f;
  for (int i = 0; i < per; ++i) { float ev = __expf(vals[i] - mx); vals[i] = ev; sum += ev; }
#pragma unroll
  for (int o = 16; o >= 1; o >>= 1) sum += __shfl_xor(sum, o, 32);
  float inv = 1.0f / sum;
  for (int i = 0; i < per; ++i) {
    int j = lane + (i << 5);
    sA[a_idx(numKT, r, j)] = f2bf(vals[i] * inv);
  }
}

// ---------------- Kernel A: per-node Feature GAT -> cat[:, :, 0:256], plus raw x -> [512:768]
__global__ void __launch_bounds__(256) k_feat_gat(
    const float* __restrict__ x, const float* __restrict__ fadj,
    const float* __restrict__ FW, const float* __restrict__ Fa,
    float* __restrict__ cat)
{
  extern __shared__ char smraw[];
  u16* sW   = (u16*)smraw;            // B-swz 8KT x 16NT : 65536
  u16* sX   = sW  + 65536;            // A-swz 4MT x 8KT  : 16384
  u16* sWh  = sX  + 16384;            // B-swz 2KT x 16NT : 16384
  u16* sAtt = sWh + 16384;            // A-swz 4MT x 2KT  : 4096
  float* sF1 = (float*)(sAtt + 4096);
  float* sF2 = sF1 + Ss;

  const int n = blockIdx.x;
  const int tid = threadIdx.x, lane = tid & 31, wv = tid >> 5;
  const float* xg = x  + (size_t)n * Ss * Ww;
  const float* Wg = FW + (size_t)n * Ww * Ww;

  stage_a_rm(sX, xg, Ss, Ww, Ww, 8, tid, 256);
  stage_b_rm(sW, Wg, Ww, Ww, Ww, 16, tid, 256);
  __syncthreads();

  const int mt = wv >> 1, ntb = (wv & 1) << 3;   // 4 MT x 16 NT over 8 waves
  {
    v8f acc[8];
#pragma unroll
    for (int j = 0; j < 8; ++j) acc[j] = vzero8();
    for (int k = 0; k < 8; ++k) {                 // K = 256
      v16bf af = load_frag(sX + ((mt * 8 + k) << 9), lane);
#pragma unroll
      for (int j = 0; j < 8; ++j) {
        v16bf bf = load_frag(sW + ((k * 16 + ntb + j) << 9), lane);
        acc[j] = WMMA_BF16(af, bf, acc[j]);
      }
    }
#pragma unroll
    for (int j = 0; j < 8; ++j)
      store_c_bswz(sWh, 16, mt * 16, (ntb + j) * 16, lane, acc[j]);
  }
  __syncthreads();

  if (tid < Ss) {
    const float* ag = Fa + (size_t)n * 2 * Ww;
    float s1 = 0.f, s2 = 0.f;
    for (int o = 0; o < Ww; ++o) {
      float wh = read_bswz(sWh, 16, tid, o);
      s1 += wh * ag[o];
      s2 += wh * ag[Ww + o];
    }
    sF1[tid] = s1; sF2[tid] = s2;
  }
  __syncthreads();

  for (int r = wv; r < Ss; r += 8)
    att_row_aswz(sAtt, 2, r, fadj + (size_t)n * Ss * Ss + r * Ss, sF1[r], sF2, 2, lane);
  __syncthreads();

  {
    v8f acc[8];
#pragma unroll
    for (int j = 0; j < 8; ++j) acc[j] = vzero8();
    for (int k = 0; k < 2; ++k) {                 // K = 64
      v16bf af = load_frag(sAtt + ((mt * 2 + k) << 9), lane);
#pragma unroll
      for (int j = 0; j < 8; ++j) {
        v16bf bf = load_frag(sWh + ((k * 16 + ntb + j) << 9), lane);
        acc[j] = WMMA_BF16(af, bf, acc[j]);
      }
    }
    float* catn = cat + (size_t)n * Ss * DIN;
#pragma unroll
    for (int j = 0; j < 8; ++j)
      store_c_g(catn + mt * 16 * DIN + (ntb + j) * 16, DIN, lane, acc[j]);
  }

  {   // raw x copy into cat[:, 512:768]
    float* catn = cat + (size_t)n * Ss * DIN;
    for (int i = tid; i < Ss * Ww; i += 256) {
      int s = i >> 8, w1 = i & 255;
      catn[s * DIN + 2 * Ww + w1] = xg[i];
    }
  }
}

// ---------------- Kernel B: per-node Time GAT (transposed) -> cat[:, :, 256:512]
__global__ void __launch_bounds__(256) k_time_gat(
    const float* __restrict__ x, const float* __restrict__ tadj,
    const float* __restrict__ TW, const float* __restrict__ Ta,
    float* __restrict__ cat)
{
  extern __shared__ char smraw[];
  u16* sXT  = (u16*)smraw;           // A-swz 16MT x 2KT : 16384  (x^T)
  u16* sWm  = sXT + 16384;           // B-swz 2KT x 4NT  : 4096
  u16* sWh  = sWm + 4096;            // B-swz 8KT x 4NT  : 16384
  u16* sAtt = sWh + 16384;           // A-swz 16MT x 8KT : 65536
  float* sF1 = (float*)(sAtt + 65536);
  float* sF2 = sF1 + Ww;

  const int n = blockIdx.x;
  const int tid = threadIdx.x, lane = tid & 31, wv = tid >> 5;
  const float* xg = x + (size_t)n * Ss * Ww;

  stage_a_tr(sXT, xg, Ww, Ss, Ww, 2, tid, 256);                  // A[w][s] = x[s][w]
  stage_b_rm(sWm, TW + (size_t)n * Ss * Ss, Ss, Ss, Ss, 4, tid, 256);
  __syncthreads();

  for (int j = 0; j < 8; ++j) {                  // 16 MT x 4 NT
    int t = wv * 8 + j, mt = t >> 2, nt = t & 3;
    v8f acc = vzero8();
    for (int k = 0; k < 2; ++k) {                // K = 64
      v16bf af = load_frag(sXT + ((mt * 2 + k) << 9), lane);
      v16bf bf = load_frag(sWm + ((k * 4 + nt) << 9), lane);
      acc = WMMA_BF16(af, bf, acc);
    }
    store_c_bswz(sWh, 4, mt * 16, nt * 16, lane, acc);
  }
  __syncthreads();

  {
    const float* ag = Ta + (size_t)n * 2 * Ss;
    float s1 = 0.f, s2 = 0.f;
    for (int o = 0; o < Ss; ++o) {
      float wh = read_bswz(sWh, 4, tid, o);
      s1 += wh * ag[o];
      s2 += wh * ag[Ss + o];
    }
    sF1[tid] = s1; sF2[tid] = s2;
  }
  __syncthreads();

  for (int r = wv; r < Ww; r += 8)
    att_row_aswz(sAtt, 8, r, tadj + (size_t)n * Ww * Ww + r * Ww, sF1[r], sF2, 8, lane);
  __syncthreads();

  float* catn = cat + (size_t)n * Ss * DIN;
  for (int j = 0; j < 8; ++j) {
    int t = wv * 8 + j, mt = t >> 2, nt = t & 3;
    v8f acc = vzero8();
    for (int k = 0; k < 8; ++k) {                // K = 256
      v16bf af = load_frag(sAtt + ((mt * 8 + k) << 9), lane);
      v16bf bf = load_frag(sWh + ((k * 4 + nt) << 9), lane);
      acc = WMMA_BF16(af, bf, acc);
    }
    // out[w][s] written transposed: cat[n][s][256+w]
    store_c_gT(catn + (nt * 16) * DIN + Ww + mt * 16, DIN, lane, acc);
  }
}

// ---------------- Kernel C: GI0[n,t,g] = cat[n,t,:] . Wih0[n,g,:] + bih0  (hoisted GRU input proj)
__global__ void __launch_bounds__(256) k_gi0(
    const float* __restrict__ cat, const float* __restrict__ Wih0,
    const float* __restrict__ bih0, float* __restrict__ GI0)
{
  extern __shared__ char smraw[];
  u16* sA = (u16*)smraw;             // A-swz 4MT x 24KT : 49152
  const int n = blockIdx.x;
  const int tid = threadIdx.x, lane = tid & 31, wv = tid >> 5;
  stage_a_rm(sA, cat + (size_t)n * Ss * DIN, Ss, DIN, DIN, 24, tid, 256);
  __syncthreads();

  const float* Wg = Wih0 + (size_t)n * Gg * DIN;
  const float* bg = bih0 + (size_t)n * Gg;
  float* og = GI0 + (size_t)n * Ss * Gg;

  for (int i = 0; i < 3; ++i) {                  // 4 MT x 6 NT over 8 waves
    int t = wv * 3 + i, mt = t / 6, nt = t % 6;
    v8f acc = vzero8();
    for (int k = 0; k < 24; ++k) {               // K = 768
      const float* Bp = Wg + (nt * 16) * DIN + k * 32;
      __builtin_prefetch(Bp + 32, 0, 0);         // global_prefetch_b8 on streamed weights
      v16bf bf = load_bT_g(Bp, DIN, lane);
      v16bf af = load_frag(sA + ((mt * 24 + k) << 9), lane);
      acc = WMMA_BF16(af, bf, acc);
    }
    store_c_bias_g(og + mt * 16 * Gg + nt * 16, Gg, bg + nt * 16, lane, acc);
  }
}

// ---------------- Kernel D: sequential 2-layer GRU recurrence.
// GI0[t] is double-buffered in LDS via GLOBAL_LOAD_ASYNC_TO_LDS_B32 (ASYNCcnt),
// prefetching step t+1 while step t's hidden-side matvecs run.
__global__ void __launch_bounds__(96) k_gru(
    const float* __restrict__ GI0, const float* __restrict__ Whh0, const float* __restrict__ bhh0,
    const float* __restrict__ Wih1, const float* __restrict__ bih1,
    const float* __restrict__ Whh1, const float* __restrict__ bhh1,
    const float* __restrict__ Hpre, float* __restrict__ gruout, float* __restrict__ outH)
{
  __shared__ float sWhh0[Gg * Hh], sWih1[Gg * Hh], sWhh1[Gg * Hh];
  __shared__ float sGIbuf[2][Gg];
  __shared__ float sGH0[Gg], sGI1[Gg], sGH1[Gg];
  __shared__ float sH0[Hh], sH1[Hh], sH0n[Hh];

  const int n = blockIdx.x;
  const int g = threadIdx.x;
  for (int i = g; i < Gg * Hh; i += 96) {
    sWhh0[i] = Whh0[(size_t)n * Gg * Hh + i];
    sWih1[i] = Wih1[(size_t)n * Gg * Hh + i];
    sWhh1[i] = Whh1[(size_t)n * Gg * Hh + i];
  }
  if (g < Hh) {
    sH0[g] = Hpre[(size_t)n * Hh + g];
    sH1[g] = Hpre[(size_t)Nn * Hh + (size_t)n * Hh + g];
  }
  float bhh0v = bhh0[(size_t)n * Gg + g];
  float bih1v = bih1[(size_t)n * Gg + g];
  float bhh1v = bhh1[(size_t)n * Gg + g];

  const float* gi_base = GI0 + (size_t)n * Ss * Gg;
  // async prefetch of GI0[t] (one b32 per lane) into LDS double buffer
  auto issue_gi = [&](int t, int buf) {
    unsigned int lds = (unsigned int)(unsigned long long)(void*)&sGIbuf[buf][g];
    unsigned long long ga = (unsigned long long)(const void*)(gi_base + (size_t)t * Gg + g);
    asm volatile("global_load_async_to_lds_b32 %0, %1, off" :: "v"(lds), "v"(ga) : "memory");
  };
  issue_gi(0, 0);                                 // ASYNCcnt = 1
  __syncthreads();

  for (int t = 0; t < Ss; ++t) {
    // hidden-side matvec for layer 0 (independent of GI0[t])
    float acc = bhh0v;
    const float* wr = sWhh0 + g * Hh;
    for (int h = 0; h < Hh; ++h) acc += wr[h] * sH0[h];
    sGH0[g] = acc;
    if (t + 1 < Ss) {
      issue_gi(t + 1, (t + 1) & 1);               // ASYNCcnt = 2
      asm volatile("s_wait_asynccnt 0x1" ::: "memory");   // step t's load complete (in-order)
    } else {
      asm volatile("s_wait_asynccnt 0x0" ::: "memory");
    }
    __syncthreads();
    const float* gi = sGIbuf[t & 1];
    if (g < Hh) {
      float r  = 1.f / (1.f + __expf(-(gi[g]      + sGH0[g])));
      float z  = 1.f / (1.f + __expf(-(gi[Hh + g] + sGH0[Hh + g])));
      float nn = tanhf(gi[2 * Hh + g] + r * sGH0[2 * Hh + g]);
      sH0n[g] = (1.f - z) * nn + z * sH0[g];
    }
    __syncthreads();
    float a1 = bih1v, a2 = bhh1v;
    const float* w1 = sWih1 + g * Hh;
    const float* w2 = sWhh1 + g * Hh;
    for (int h = 0; h < Hh; ++h) { a1 += w1[h] * sH0n[h]; a2 += w2[h] * sH1[h]; }
    sGI1[g] = a1; sGH1[g] = a2;
    __syncthreads();
    if (g < Hh) {
      float r  = 1.f / (1.f + __expf(-(sGI1[g]      + sGH1[g])));
      float z  = 1.f / (1.f + __expf(-(sGI1[Hh + g] + sGH1[Hh + g])));
      float nn = tanhf(sGI1[2 * Hh + g] + r * sGH1[2 * Hh + g]);
      float h1n = (1.f - z) * nn + z * sH1[g];
      gruout[(size_t)n * Ss * Hh + t * Hh + g] = h1n;
      sH0[g] = sH0n[g];
      sH1[g] = h1n;
    }
    __syncthreads();
  }
  if (g < Hh) {                                   // cat_H output [2,N,H]
    outH[(size_t)n * Hh + g]                   = sH0[g];
    outH[(size_t)Nn * Hh + (size_t)n * Hh + g] = sH1[g];
  }
}

// ---------------- Kernel E: per-node FC stack (WMMA 64x32x32 twice + matvec) -> ct[N,S]
__global__ void __launch_bounds__(256) k_fc(
    const float* __restrict__ gruout,
    const float* __restrict__ fc1w, const float* __restrict__ fc1b,
    const float* __restrict__ fc2w, const float* __restrict__ fc2b,
    const float* __restrict__ fc3w, const float* __restrict__ fc3b,
    float* __restrict__ ct)
{
  __shared__ u16 sG[2048], sB[1024], sG2[2048], sG3[2048];   // A-swz / B-swz tiles
  const int n = blockIdx.x;
  const int tid = threadIdx.x, lane = tid & 31, wv = tid >> 5;

  stage_a_rm(sG, gruout + (size_t)n * Ss * Hh, Ss, Hh, Hh, 1, tid, 256);
  stage_b_tr(sB, fc1w + (size_t)n * Hh * Hh, Hh, Hh, Hh, 2, tid, 256);  // B[h][kc]=w[kc][h]
  __syncthreads();

  const int mt = wv >> 1, nt = wv & 1;            // 4 MT x 2 NT, 1 tile per wave
  {
    v16bf af = load_frag(sG + (mt << 9), lane);
    v16bf bf = load_frag(sB + (nt << 9), lane);
    v8f acc = vzero8();
    acc = WMMA_BF16(af, bf, acc);
    store_c_aswz_relu_bias(sG2, 1, mt * 16, nt * 16, fc1b + (size_t)n * Hh + nt * 16, lane, acc);
  }
  __syncthreads();
  stage_b_tr(sB, fc2w + (size_t)n * Hh * Hh, Hh, Hh, Hh, 2, tid, 256);
  __syncthreads();
  {
    v16bf af = load_frag(sG2 + (mt << 9), lane);
    v16bf bf = load_frag(sB + (nt << 9), lane);
    v8f acc = vzero8();
    acc = WMMA_BF16(af, bf, acc);
    store_c_aswz_relu_bias(sG3, 1, mt * 16, nt * 16, fc2b + (size_t)n * Hh + nt * 16, lane, acc);
  }
  __syncthreads();
  if (tid < Ss) {
    float acc = fc3b[n];
    for (int h = 0; h < Hh; ++h) acc += read_aswz(sG3, 1, tid, h) * fc3w[(size_t)n * Hh + h];
    ct[(size_t)n * Ss + tid] = acc;
  }
}

// ---------------- Kernel F: three global output GATs over ct[256,64], one workgroup
__device__ void out_gat_stage(const u16* sA, int K, const float* Wg, const float* ag,
                              const float* adjg, u16* sWm, u16* sWh, u16* sAtt,
                              float* sF1, float* sF2, u16* outA /*A-swz dest*/, int outK0,
                              float* outG)
{
  const int tid = threadIdx.x, lane = tid & 31, wv = tid >> 5;
  const int numKTa = K >> 5;
  stage_b_rm(sWm, Wg, K, Ss, Ss, 4, tid, 512);
  __syncthreads();
  for (int j = 0; j < 4; ++j) {                   // 16 MT x 4 NT over 16 waves
    int t = wv * 4 + j, mt = t >> 2, nt = t & 3;
    v8f acc = vzero8();
    for (int k = 0; k < numKTa; ++k) {
      v16bf af = load_frag(sA + ((mt * numKTa + k) << 9), lane);
      v16bf bf = load_frag(sWm + ((k * 4 + nt) << 9), lane);
      acc = WMMA_BF16(af, bf, acc);
    }
    store_c_bswz(sWh, 4, mt * 16, nt * 16, lane, acc);
  }
  __syncthreads();
  if (tid < Nn) {
    float s1 = 0.f, s2 = 0.f;
    for (int o = 0; o < Ss; ++o) {
      float wh = read_bswz(sWh, 4, tid, o);
      s1 += wh * ag[o];
      s2 += wh * ag[Ss + o];
    }
    sF1[tid] = s1; sF2[tid] = s2;
  }
  __syncthreads();
  for (int r = wv; r < Nn; r += 16)
    att_row_aswz(sAtt, 8, r, adjg + r * Nn, sF1[r], sF2, 8, lane);
  __syncthreads();
  for (int j = 0; j < 4; ++j) {
    int t = wv * 4 + j, mt = t >> 2, nt = t & 3;
    v8f acc = vzero8();
    for (int k = 0; k < 8; ++k) {                 // K = 256
      v16bf af = load_frag(sAtt + ((mt * 8 + k) << 9), lane);
      v16bf bf = load_frag(sWh + ((k * 4 + nt) << 9), lane);
      acc = WMMA_BF16(af, bf, acc);
    }
    if (outA) store_c_aswz(outA, 4, mt * 16, outK0 + nt * 16, lane, acc);
    else      store_c_g(outG + mt * 16 * Ss + nt * 16, Ss, lane, acc);
  }
  __syncthreads();
}

__global__ void __launch_bounds__(512) k_out_gat(
    const float* __restrict__ ct, const float* __restrict__ adj,
    const float* __restrict__ oW,  const float* __restrict__ oa,
    const float* __restrict__ o1W, const float* __restrict__ o1a,
    const float* __restrict__ o2W, const float* __restrict__ o2a,
    float* __restrict__ finalout)
{
  extern __shared__ char smraw[];
  u16* sCt   = (u16*)smraw;            // A-swz 16MT x 2KT : 16384
  u16* sWm   = sCt   + 16384;          // B-swz up to 4KT x 4NT : 8192
  u16* sWh   = sWm   + 8192;           // B-swz 8KT x 4NT : 16384
  u16* sAtt  = sWh   + 16384;          // A-swz 16MT x 8KT : 65536
  u16* sCat2 = sAtt  + 65536;          // A-swz 16MT x 4KT : 32768
  float* sF1 = (float*)(sCat2 + 32768);
  float* sF2 = sF1 + Nn;

  const int tid = threadIdx.x;
  stage_a_rm(sCt, ct, Nn, Ss, Ss, 2, tid, 512);
  __syncthreads();

  out_gat_stage(sCt,   Ss,  oW,  oa,  adj, sWm, sWh, sAtt, sF1, sF2, sCat2, 0,  nullptr);
  out_gat_stage(sCt,   Ss,  o1W, o1a, adj, sWm, sWh, sAtt, sF1, sF2, sCat2, Ss, nullptr);
  out_gat_stage(sCat2, 128, o2W, o2a, adj, sWm, sWh, sAtt, sF1, sF2, nullptr, 0, finalout);
}

// ---------------- Host entry ----------------
extern "C" void kernel_launch(void* const* d_in, const int* in_sizes, int n_in,
                              void* d_out, int out_size, void* d_ws, size_t ws_size,
                              hipStream_t stream) {
  const float* x    = (const float*)d_in[0];
  const float* fadj = (const float*)d_in[1];
  const float* tadj = (const float*)d_in[2];
  const float* adj  = (const float*)d_in[3];
  const float* Hpre = (const float*)d_in[4];
  const float* FW   = (const float*)d_in[5];
  const float* Fa   = (const float*)d_in[6];
  const float* TW   = (const float*)d_in[7];
  const float* Ta   = (const float*)d_in[8];
  const float* Wih0 = (const float*)d_in[9];
  const float* Whh0 = (const float*)d_in[10];
  const float* bih0 = (const float*)d_in[11];
  const float* bhh0 = (const float*)d_in[12];
  const float* Wih1 = (const float*)d_in[13];
  const float* Whh1 = (const float*)d_in[14];
  const float* bih1 = (const float*)d_in[15];
  const float* bhh1 = (const float*)d_in[16];
  const float* fc1w = (const float*)d_in[17];
  const float* fc1b = (const float*)d_in[18];
  const float* fc2w = (const float*)d_in[19];
  const float* fc2b = (const float*)d_in[20];
  const float* fc3w = (const float*)d_in[21];
  const float* fc3b = (const float*)d_in[22];
  const float* oW   = (const float*)d_in[23];
  const float* oa   = (const float*)d_in[24];
  const float* o1W  = (const float*)d_in[25];
  const float* o1a  = (const float*)d_in[26];
  const float* o2W  = (const float*)d_in[27];
  const float* o2a  = (const float*)d_in[28];

  float* ws     = (float*)d_ws;
  float* cat    = ws;                                   // N*S*3W
  float* GI0    = cat    + (size_t)Nn * Ss * DIN;       // N*S*96
  float* gruout = GI0    + (size_t)Nn * Ss * Gg;        // N*S*H
  float* ctb    = gruout + (size_t)Nn * Ss * Hh;        // N*S

  float* fout = (float*)d_out;                          // finalout [N,S]
  float* outH = fout + (size_t)Nn * Ss;                 // cat_H [2,N,H]

  size_t ldsA = (size_t)(65536 + 16384 + 16384 + 4096) * 2 + (size_t)(Ss + Ss) * 4;
  k_feat_gat<<<Nn, 256, ldsA, stream>>>(x, fadj, FW, Fa, cat);

  size_t ldsB = (size_t)(16384 + 4096 + 16384 + 65536) * 2 + (size_t)(Ww + Ww) * 4;
  k_time_gat<<<Nn, 256, ldsB, stream>>>(x, tadj, TW, Ta, cat);

  size_t ldsC = (size_t)49152 * 2;
  k_gi0<<<Nn, 256, ldsC, stream>>>(cat, Wih0, bih0, GI0);

  k_gru<<<Nn, 96, 0, stream>>>(GI0, Whh0, bhh0, Wih1, bih1, Whh1, bhh1, Hpre, gruout, outH);

  k_fc<<<Nn, 256, 0, stream>>>(gruout, fc1w, fc1b, fc2w, fc2b, fc3w, fc3b, ctb);

  size_t ldsF = (size_t)(16384 + 8192 + 16384 + 65536 + 32768) * 2 + (size_t)(Nn + Nn) * 4;
  k_out_gat<<<1, 512, ldsF, stream>>>(ctb, adj, oW, oa, o1W, o1a, o2W, o2a, fout);
}
